// STCN_39539468927574
// MI455X (gfx1250) — matrix operand order repaired and checked by reading, not compiled, not run on hardware
//
#include <hip/hip_runtime.h>

// ---------------------------------------------------------------------------
// Fused STCN affinity+readout for MI455X (gfx1250, wave32, WMMA).
//   sim  = (-mk^3.qe + 2 mk.(qk qe) - b_sq) * ms / 8   -> bf16 WMMA, K=32
//   aff  = softmax over the 4 heads (per (n,q) point)  -> fp32 VALU exp
//   mem  = mv @ aff                                    -> bf16 WMMA, fp32 acc
// mv is pre-converted to zero-padded bf16 when workspace allows (fast path);
// otherwise converted inline in the hot loop (fallback path).
// ---------------------------------------------------------------------------

typedef __attribute__((ext_vector_type(16))) __bf16 v16bf;
typedef __attribute__((ext_vector_type(8)))  float  v8f;

#define B_      2
#define CK_     64
#define CV_     512
#define HW_     1620            // 30*54
#define THW_    6480            // 4*1620
#define NPAD_   6496            // THW padded to 32-multiple
#define HEADS_  4
#define CH_     16              // CK / HEADS
#define CVH_    128             // CV / HEADS
#define QT_     102             // ceil(1620/16)
#define NCHUNK_ 203             // NPAD/32
#define NT16_   406             // NPAD/16 16-row subtiles

static __device__ __forceinline__ unsigned short f2bf(float f) {
  unsigned u = __float_as_uint(f);
  u += 0x7FFFu + ((u >> 16) & 1u);          // round-to-nearest-even
  return (unsigned short)(u >> 16);
}

union V16U {
  v16bf v;
  unsigned short u[16];
  uint4 q[2];
};

// --------------------------------------------------------------------------
// b_sq[b,h,q] = sum_c qe * qk^3
// --------------------------------------------------------------------------
__global__ void prep_bsq_kernel(const float* __restrict__ qk,
                                const float* __restrict__ qe,
                                float* __restrict__ bsq) {
  int idx = blockIdx.x * blockDim.x + threadIdx.x;
  if (idx >= B_ * HEADS_ * HW_) return;
  int q  = idx % HW_;
  int bh = idx / HW_;                       // b*4 + h
  float s = 0.0f;
  for (int c = 0; c < CH_; ++c) {
    size_t off = ((size_t)bh * CH_ + c) * HW_ + q;
    float k = qk[off];
    s += qe[off] * k * k * k;
  }
  bsq[idx] = s;
}

// --------------------------------------------------------------------------
// Pack sim B-matrix (32k x 16q) per (b,h,qtile), in WMMA B register layout:
//   lane l: column q=l&15; c = (l>>4)*8 + (i&7)
//   elems 0-7 -> -qe[c];  elems 8-15 -> 2*qk*qe[c]
// --------------------------------------------------------------------------
__global__ __launch_bounds__(128) void prep_bsim_kernel(
    const float* __restrict__ qk, const float* __restrict__ qe,
    unsigned short* __restrict__ bsim) {
  int widx = blockIdx.x * 4 + (threadIdx.x >> 5);   // 0..815
  int lane = threadIdx.x & 31;
  if (widx >= B_ * HEADS_ * QT_) return;
  int qt = widx % QT_;
  int bh = widx / QT_;
  int q  = qt * 16 + (lane & 15);
  int side = lane >> 4;
  V16U v;
  for (int i = 0; i < 16; ++i) {
    float val = 0.0f;
    if (q < HW_) {
      size_t off = ((size_t)bh * CH_ + (side << 3) + (i & 7)) * HW_ + q;
      float e = qe[off];
      val = (i >> 3) ? 2.0f * qk[off] * e : -e;
    }
    v.u[i] = f2bf(val);
  }
  uint4* dst = reinterpret_cast<uint4*>(bsim + (size_t)widx * 512 + lane * 16);
  dst[0] = v.q[0];
  dst[1] = v.q[1];
}

// --------------------------------------------------------------------------
// Pack sim A-matrix (16n x 32k) per (b,h,n16-tile), WMMA A register layout:
//   lane l: row M = l&15; klo = (l>>4)*8
//   elems 0-7  -> K = klo+j    = mk^3[c=klo+j]
//   elems 8-15 -> K = 16+klo+j = mk  [c=klo+j]
// n >= 6480 padded with zeros.
// --------------------------------------------------------------------------
__global__ __launch_bounds__(128) void prep_mkA_kernel(
    const float* __restrict__ mk, unsigned short* __restrict__ mkA) {
  int widx = blockIdx.x * 4 + (threadIdx.x >> 5);   // 0..3247
  int lane = threadIdx.x & 31;
  if (widx >= B_ * HEADS_ * NT16_) return;
  int t   = widx % NT16_;
  int bh  = widx / NT16_;
  int n   = t * 16 + (lane & 15);
  int klo = (lane >> 4) << 3;
  V16U v;
  for (int j = 0; j < 8; ++j) {
    float m = (n < THW_) ? mk[((size_t)bh * CH_ + klo + j) * THW_ + n] : 0.0f;
    v.u[j]     = f2bf(m * m * m);
    v.u[8 + j] = f2bf(m);
  }
  uint4* dst = reinterpret_cast<uint4*>(mkA + (size_t)widx * 512 + lane * 16);
  dst[0] = v.q[0];
  dst[1] = v.q[1];
}

// --------------------------------------------------------------------------
// mv fp32 -> bf16, n padded 6480 -> 6496 with zeros. One thread per 8 elems.
// --------------------------------------------------------------------------
__global__ void prep_mv_kernel(const float* __restrict__ mv,
                               unsigned short* __restrict__ mvbf) {
  const int NG = NPAD_ / 8;                       // 812 groups per row
  int idx = blockIdx.x * blockDim.x + threadIdx.x;
  if (idx >= B_ * CV_ * NG) return;
  int g   = idx % NG;
  int row = idx / NG;                             // b*512 + cv
  int n   = g * 8;
  union { unsigned short u[8]; uint4 q; } pk;
  if (n + 8 <= THW_) {                            // THW%8==0: no straddle
    const float4* p =
        reinterpret_cast<const float4*>(mv + (size_t)row * THW_ + n);
    float4 a = p[0], c = p[1];
    const float f[8] = {a.x, a.y, a.z, a.w, c.x, c.y, c.z, c.w};
#pragma unroll
    for (int i = 0; i < 8; ++i) pk.u[i] = f2bf(f[i]);
  } else {
#pragma unroll
    for (int i = 0; i < 8; ++i) pk.u[i] = 0;
  }
  reinterpret_cast<uint4*>(mvbf + (size_t)row * NPAD_ + n)[0] = pk.q;
}

// --------------------------------------------------------------------------
// out[:, 512:1024, :, :] = qv
// --------------------------------------------------------------------------
__global__ void copy_qv_kernel(const float* __restrict__ qv,
                               float* __restrict__ out) {
  int idx = blockIdx.x * blockDim.x + threadIdx.x;
  const int N = B_ * CV_ * HW_;
  if (idx >= N) return;
  int b   = idx / (CV_ * HW_);
  int rem = idx - b * (CV_ * HW_);
  out[(size_t)b * (2 * CV_) * HW_ + (size_t)CV_ * HW_ + rem] = qv[idx];
}

// --------------------------------------------------------------------------
// Main fused kernel. grid = (102 qtiles, B, 2 cv-halves), 128 threads (4 waves).
// Wave w: sim head w (phase A), cv slice (half*4 + w) of every head (phase C).
// MVBF16: mv pre-converted/padded bf16 (fast), else inline fp32->bf16.
// --------------------------------------------------------------------------
template <bool MVBF16>
__global__ __launch_bounds__(128) void stcn_main_kernel(
    const unsigned short* __restrict__ mkA,
    const unsigned short* __restrict__ bsim,
    const float* __restrict__ bsq,
    const float* __restrict__ ms,
    const float* __restrict__ mv,
    const unsigned short* __restrict__ mvbf,
    float* __restrict__ out) {
  __shared__ __align__(16) float          simLDS[32 * 16 * HEADS_]; // [n32][q16][h]
  __shared__ __align__(16) unsigned short affLDS[HEADS_ * 512];     // [h][lane][elem]

  const int tid  = threadIdx.x;
  const int wave = tid >> 5;
  const int lane = tid & 31;
  const int qt   = blockIdx.x;
  const int b    = blockIdx.y;
  const int half = blockIdx.z;
  const int q0   = qt * 16;
  const int qcol = lane & 15;
  const int hi   = lane >> 4;
  const int klo  = hi << 3;
  const int hh   = wave;               // this wave's sim head
  const int sl   = half * 4 + wave;    // cv slice 0..7

  // Hoisted per-wave sim B operand (chunk-invariant) and b_sq.
  V16U bop;
  {
    const uint4* p = reinterpret_cast<const uint4*>(
        bsim + ((size_t)((b * HEADS_ + hh) * QT_ + qt)) * 512 + lane * 16);
    bop.q[0] = p[0];
    bop.q[1] = p[1];
  }
  const int   q  = q0 + qcol;
  const float bq = (q < HW_) ? bsq[(b * HEADS_ + hh) * HW_ + q] : 0.0f;

  v8f acc[HEADS_];
  for (int h2 = 0; h2 < HEADS_; ++h2) acc[h2] = {};

  for (int chunk = 0; chunk < NCHUNK_; ++chunk) {
    // ---- Phase A: sim tiles (one WMMA per 16n x 16q tile) ----
    for (int s = 0; s < 2; ++s) {
      const int n16 = chunk * 32 + s * 16;
      const int t   = chunk * 2 + s;
      V16U aop;
      const uint4* ap = reinterpret_cast<const uint4*>(
          mkA + ((size_t)((b * HEADS_ + hh) * NT16_ + t)) * 512 + lane * 16);
      aop.q[0] = ap[0];
      aop.q[1] = ap[1];
      v8f z = {};
      v8f sim = __builtin_amdgcn_wmma_f32_16x16x32_bf16(
          false, aop.v, false, bop.v, (short)0, z, false, false);
      // ms for this lane's 8 rows; group fully in or fully out of range.
      const int mbase = n16 + (hi << 3);
      float4 m0 = {0, 0, 0, 0}, m1 = {0, 0, 0, 0};
      if (mbase < THW_) {
        const float4* mp = reinterpret_cast<const float4*>(ms + b * THW_ + mbase);
        m0 = mp[0];
        m1 = mp[1];
      }
      const float msv[8] = {m0.x, m0.y, m0.z, m0.w, m1.x, m1.y, m1.z, m1.w};
      float* sp = &simLDS[((s * 16 + (hi << 3)) * 16 + qcol) * HEADS_ + hh];
#pragma unroll
      for (int r = 0; r < 8; ++r)
        sp[r * 16 * HEADS_] = (sim[r] - bq) * (msv[r] * 0.125f);
    }
    __syncthreads();

    // ---- Phase B: head-softmax, scatter bf16 aff into readout-B layout ----
#pragma unroll
    for (int j = 0; j < 4; ++j) {
      const int p  = tid + 128 * j;       // p = K*16 + qc, K in [0,32)
      const int K  = p >> 4;
      const int qc = p & 15;
      const float4 sv = *reinterpret_cast<const float4*>(&simLDS[p * HEADS_]);
      float m  = fmaxf(fmaxf(sv.x, sv.y), fmaxf(sv.z, sv.w));
      float e0 = __expf(sv.x - m), e1 = __expf(sv.y - m);
      float e2 = __expf(sv.z - m), e3 = __expf(sv.w - m);
      float rinv = 1.0f / (e0 + e1 + e2 + e3);
      const int lt = qc + (((K >> 3) & 1) << 4);
      unsigned short* dst = &affLDS[lt * 16 + (K & 7) + ((K >> 4) << 3)];
      dst[0 * 512] = f2bf(e0 * rinv);
      dst[1 * 512] = f2bf(e1 * rinv);
      dst[2 * 512] = f2bf(e2 * rinv);
      dst[3 * 512] = f2bf(e3 * rinv);
    }
    __syncthreads();

    // ---- Phase C: readout, one WMMA per head ----
    const int Mrow = lane & 15;
#pragma unroll
    for (int h2 = 0; h2 < HEADS_; ++h2) {
      const int ch = h2 * CVH_ + sl * 16 + Mrow;
      V16U am;
      if (MVBF16) {
        const uint4* rp = reinterpret_cast<const uint4*>(
            mvbf + ((size_t)(b * CV_ + ch)) * NPAD_ + chunk * 32 + klo);
        am.q[0] = rp[0];   // elems 0-7  : K = klo..klo+7
        am.q[1] = rp[2];   // elems 8-15 : K = 16+klo.. (+16 halves = +32B)
        if (chunk + 1 < NCHUNK_)
          __builtin_prefetch((const void*)(rp + 4), 0, 1);  // next chunk row
      } else {
        const bool g1ok = (chunk * 32 + 16 + klo + 8) <= THW_;
        const float* rowp = mv + ((size_t)(b * CV_ + ch)) * THW_ + chunk * 32;
        float4 f0 = *reinterpret_cast<const float4*>(rowp + klo);
        float4 f1 = *reinterpret_cast<const float4*>(rowp + klo + 4);
        float4 f2 = {0, 0, 0, 0}, f3 = {0, 0, 0, 0};
        if (g1ok) {
          f2 = *reinterpret_cast<const float4*>(rowp + 16 + klo);
          f3 = *reinterpret_cast<const float4*>(rowp + 20 + klo);
        }
        const float fa[16] = {f0.x, f0.y, f0.z, f0.w, f1.x, f1.y, f1.z, f1.w,
                              f2.x, f2.y, f2.z, f2.w, f3.x, f3.y, f3.z, f3.w};
#pragma unroll
        for (int i = 0; i < 16; ++i) am.u[i] = f2bf(fa[i]);
      }
      const uint4* bp =
          reinterpret_cast<const uint4*>(&affLDS[h2 * 512 + lane * 16]);
      V16U bm;
      bm.q[0] = bp[0];
      bm.q[1] = bp[1];
      acc[h2] = __builtin_amdgcn_wmma_f32_16x16x32_bf16(
          false, am.v, false, bm.v, (short)0, acc[h2], false, false);
    }
    // No barrier needed here: next phase-A writes simLDS (disjoint from
    // affLDS) and phase-B writes are fenced by the first barrier above.
  }

  // ---- Store: C layout row M = r + 8*(lane>>4), col N = lane&15 ----
  if (q < HW_) {
#pragma unroll
    for (int h2 = 0; h2 < HEADS_; ++h2) {
#pragma unroll
      for (int r = 0; r < 8; ++r) {
        const int ch = h2 * CVH_ + sl * 16 + r + (hi << 3);
        out[((size_t)(b * 2 * CV_ + ch)) * HW_ + q] = acc[h2][r];
      }
    }
  }
}

// --------------------------------------------------------------------------
extern "C" void kernel_launch(void* const* d_in, const int* in_sizes, int n_in,
                              void* d_out, int out_size, void* d_ws,
                              size_t ws_size, hipStream_t stream) {
  const float* mk = (const float*)d_in[0];
  const float* qk = (const float*)d_in[1];
  const float* ms = (const float*)d_in[2];
  const float* qe = (const float*)d_in[3];
  const float* mv = (const float*)d_in[4];
  const float* qv = (const float*)d_in[5];
  float* out = (float*)d_out;

  // Workspace layout: bsq | bsim | mkA (~4.2 MB) [| mvbf (~13.3 MB)].
  char* base = (char*)d_ws;
  size_t o = 0;
  float* bsq = (float*)(base + o);
  o += (size_t)B_ * HEADS_ * HW_ * sizeof(float);
  o = (o + 255) & ~(size_t)255;
  unsigned short* bsim = (unsigned short*)(base + o);
  o += (size_t)B_ * HEADS_ * QT_ * 512 * sizeof(unsigned short);
  o = (o + 255) & ~(size_t)255;
  unsigned short* mkA = (unsigned short*)(base + o);
  o += (size_t)B_ * HEADS_ * NT16_ * 512 * sizeof(unsigned short);
  o = (o + 255) & ~(size_t)255;
  unsigned short* mvbf = (unsigned short*)(base + o);
  const size_t need_full = o + (size_t)B_ * CV_ * NPAD_ * sizeof(unsigned short);
  const bool use_bf = (ws_size >= need_full);   // deterministic per harness

  prep_bsq_kernel<<<(B_ * HEADS_ * HW_ + 255) / 256, 256, 0, stream>>>(qk, qe,
                                                                       bsq);
  prep_bsim_kernel<<<(B_ * HEADS_ * QT_ + 3) / 4, 128, 0, stream>>>(qk, qe,
                                                                    bsim);
  prep_mkA_kernel<<<(B_ * HEADS_ * NT16_ + 3) / 4, 128, 0, stream>>>(mk, mkA);
  copy_qv_kernel<<<(B_ * CV_ * HW_ + 255) / 256, 256, 0, stream>>>(qv, out);

  dim3 grid(QT_, B_, 2);
  if (use_bf) {
    prep_mv_kernel<<<(B_ * CV_ * (NPAD_ / 8) + 255) / 256, 256, 0, stream>>>(
        mv, mvbf);
    stcn_main_kernel<true><<<grid, 128, 0, stream>>>(mkA, bsim, bsq, ms, mv,
                                                     mvbf, out);
  } else {
    stcn_main_kernel<false><<<grid, 128, 0, stream>>>(mkA, bsim, bsq, ms, mv,
                                                      mvbf, out);
  }
}
